// CrossAttention_23218593202524
// MI455X (gfx1250) — matrix operand order repaired and checked by reading, not compiled, not run on hardware
//
#include <hip/hip_runtime.h>
#include <hip/hip_bf16.h>

typedef __bf16 bf16_t;
typedef __attribute__((ext_vector_type(16))) __bf16 v16bf;
typedef __attribute__((ext_vector_type(8)))  __bf16 v8bf;
typedef __attribute__((ext_vector_type(8)))  float  v8f;

static __device__ __forceinline__ v16bf cat8(v8bf lo, v8bf hi) {
  return __builtin_shufflevector(lo, hi, 0,1,2,3,4,5,6,7,8,9,10,11,12,13,14,15);
}

static __device__ __forceinline__ v8f wmma_bf16(v16bf a, v16bf b, v8f c) {
  // D = A(16x32 bf16) * B(32x16 bf16) + C(16x16 f32)
  return __builtin_amdgcn_wmma_f32_16x16x32_bf16(false, a, false, b, (short)0, c, false, false);
}

// ---------------------------------------------------------------------------
// Generic GEMM:  C[M,N] = A[M,K] * W[K,N] (+bias), A fp32 or bf16, W fp32.
// Inputs converted to bf16, accumulate fp32 in v_wmma_f32_16x16x32_bf16.
// Block tile 128x128, K-step 32, 256 threads = 8 waves (2x4), wave tile 64x32.
// ---------------------------------------------------------------------------
#define BM 128
#define BN 128
#define BK 32
#define LDA 40   // padded LDS row (bf16): 80B rows -> 16B aligned frag loads
#define LDB 40

template <typename TA>
__global__ __launch_bounds__(256)
void gemm_bf16_wmma(const TA* __restrict__ A, const float* __restrict__ W,
                    const float* __restrict__ bias,
                    float* __restrict__ Cf, bf16_t* __restrict__ Cb,
                    int M, int N, int K)
{
  __shared__ __align__(16) bf16_t As[BM * LDA];     // [row][k]
  __shared__ __align__(16) bf16_t Bs[BN * LDB];     // transposed: [n][k]

  const int tid  = threadIdx.x;
  const int lane = tid & 31;
  const int wave = tid >> 5;
  const int llo  = lane & 15;
  const int lhi  = lane >> 4;
  const int bm   = blockIdx.y * BM;
  const int bn   = blockIdx.x * BN;
  const int wm   = (wave >> 2) * 64;     // wave rows (4 tiles of 16)
  const int wn   = (wave & 3) * 32;      // wave cols (2 tiles of 16)

  v8f acc[4][2] = {};

  for (int k0 = 0; k0 < K; k0 += BK) {
    // ---- stage A tile 128x32 -> LDS (convert to bf16), 4 elems/thread/pass
    #pragma unroll
    for (int it = 0; it < 4; ++it) {
      int idx = tid * 4 + it * 1024;     // element in 128x32 tile
      int r = idx >> 5, c = idx & 31;
      bf16_t v0 = (bf16_t)0.f, v1 = v0, v2 = v0, v3 = v0;
      int gr = bm + r;
      if (gr < M) {
        const TA* p = A + (size_t)gr * K + k0 + c;
        v0 = (bf16_t)p[0]; v1 = (bf16_t)p[1]; v2 = (bf16_t)p[2]; v3 = (bf16_t)p[3];
      }
      bf16_t* q = &As[r * LDA + c];
      q[0] = v0; q[1] = v1; q[2] = v2; q[3] = v3;
    }
    // ---- stage W tile 32x128 -> LDS transposed [n][k]
    #pragma unroll
    for (int it = 0; it < 4; ++it) {
      int idx = tid * 4 + it * 1024;
      int kk = idx >> 7, nn = idx & 127;
      const float* p = W + (size_t)(k0 + kk) * N + bn + nn;
      Bs[(nn + 0) * LDB + kk] = (bf16_t)p[0];
      Bs[(nn + 1) * LDB + kk] = (bf16_t)p[1];
      Bs[(nn + 2) * LDB + kk] = (bf16_t)p[2];
      Bs[(nn + 3) * LDB + kk] = (bf16_t)p[3];
    }
    __syncthreads();

    // ---- fragments + 8 WMMAs
    v16bf afr[4], bfr[2];
    #pragma unroll
    for (int mi = 0; mi < 4; ++mi) {
      const bf16_t* pa = &As[(wm + mi * 16 + llo) * LDA + (lhi << 3)];
      afr[mi] = cat8(*(const v8bf*)pa, *(const v8bf*)(pa + 16));
    }
    #pragma unroll
    for (int ni = 0; ni < 2; ++ni) {
      const bf16_t* pb = &Bs[(wn + ni * 16 + llo) * LDB + (lhi << 3)];
      bfr[ni] = cat8(*(const v8bf*)pb, *(const v8bf*)(pb + 16));
    }
    #pragma unroll
    for (int mi = 0; mi < 4; ++mi)
      #pragma unroll
      for (int ni = 0; ni < 2; ++ni)
        acc[mi][ni] = wmma_bf16(afr[mi], bfr[ni], acc[mi][ni]);
    __syncthreads();
  }

  // ---- epilogue (C layout: VGPR j -> row j+8*lhi, col = llo)
  #pragma unroll
  for (int mi = 0; mi < 4; ++mi) {
    #pragma unroll
    for (int ni = 0; ni < 2; ++ni) {
      int gc = bn + wn + ni * 16 + llo;
      float bv = bias ? bias[gc] : 0.0f;
      #pragma unroll
      for (int j = 0; j < 8; ++j) {
        int gr = bm + wm + mi * 16 + j + (lhi << 3);
        if (gr < M) {
          float val = acc[mi][ni][j] + bv;
          if (Cf) Cf[(size_t)gr * N + gc] = val;
          else    Cb[(size_t)gr * N + gc] = (bf16_t)val;
        }
      }
    }
  }
}

// ---------------------------------------------------------------------------
// Fused attention: one block = (b, h, 128 query rows). 8 waves x 16 rows.
// sim = Q K^T * scale -> masked softmax (77 keys, padded 80) -> P V.
// Q/K/V are bf16 in workspace; O may alias Q (block reads before it writes).
// ---------------------------------------------------------------------------
#define NKP   80        // padded key count (5 tiles of 16)
#define LDK   72        // K LDS row stride (bf16), 144B
#define LDVT  88        // V^T LDS row stride, 176B
#define LDP   104       // P LDS row stride (96 keys used), 208B

__global__ __launch_bounds__(256)
void attn_fused(const bf16_t* __restrict__ Q, const bf16_t* __restrict__ Kp,
                const bf16_t* __restrict__ Vp, bf16_t* __restrict__ O,
                int Nq, int Nk)
{
  __shared__ __align__(16) bf16_t Ks[NKP * LDK];      // [key][d]
  __shared__ __align__(16) bf16_t Vt[64 * LDVT];      // [d][key]
  __shared__ __align__(16) bf16_t Ps[8][16 * LDP];    // per-wave P [row][key]

  const int tid  = threadIdx.x;
  const int lane = tid & 31;
  const int wave = tid >> 5;
  const int llo  = lane & 15;
  const int lhi  = lane >> 4;
  const int h    = blockIdx.y;
  const int b    = blockIdx.z;
  const int q0   = blockIdx.x * 128 + wave * 16;

  // ---- stage K head tile [80x64] (zero-padded keys)
  for (int it = 0; it < 20; ++it) {
    int idx = tid + it * 256;                 // 5120 = 80*64
    int key = idx >> 6, d = idx & 63;
    bf16_t v = (bf16_t)0.f;
    if (key < Nk) v = Kp[((size_t)(b * Nk + key)) * 1024 + h * 64 + d];
    Ks[key * LDK + d] = v;
  }
  // ---- stage V^T head tile [64 x 80]
  for (int it = 0; it < 20; ++it) {
    int idx = tid + it * 256;                 // 5120 = 64*80
    int d = idx / NKP, key = idx % NKP;
    bf16_t v = (bf16_t)0.f;
    if (key < Nk) v = Vp[((size_t)(b * Nk + key)) * 1024 + h * 64 + d];
    Vt[d * LDVT + key] = v;
  }
  // ---- zero P pad columns 80..95 (keys beyond padded range)
  for (int it = 0; it < 8; ++it) {
    int idx = tid + it * 256;                 // 2048 = 8*16*16
    int w = idx >> 8, rr = (idx >> 4) & 15, cc = idx & 15;
    Ps[w][rr * LDP + 80 + cc] = (bf16_t)0.f;
  }
  __syncthreads();

  // ---- sim = Q K^T : 5 N-tiles x 2 K-steps of wmma
  v8f s[5] = {};
  const bf16_t* qbase = Q + ((size_t)(b * Nq) + q0 + llo) * 1024 + h * 64;
  #pragma unroll
  for (int ks = 0; ks < 2; ++ks) {
    const bf16_t* pq = qbase + ks * 32 + (lhi << 3);
    v16bf qf = cat8(*(const v8bf*)pq, *(const v8bf*)(pq + 16));
    #pragma unroll
    for (int nt = 0; nt < 5; ++nt) {
      const bf16_t* pk = &Ks[(nt * 16 + llo) * LDK + ks * 32 + (lhi << 3)];
      v16bf kf = cat8(*(const v8bf*)pk, *(const v8bf*)(pk + 16));
      s[nt] = wmma_bf16(qf, kf, s[nt]);
    }
  }

  // ---- masked softmax over 77 keys; rows live in half-waves (16 lanes)
  const float scale = 0.125f;                 // 64^-0.5
  #pragma unroll
  for (int j = 0; j < 8; ++j) {
    float vals[5];
    float m = -3.0e38f;
    #pragma unroll
    for (int nt = 0; nt < 5; ++nt) {
      float v = s[nt][j] * scale;
      bool ok = (nt * 16 + llo) < 77;
      vals[nt] = ok ? v : -3.0e38f;
      m = fmaxf(m, vals[nt]);
    }
    #pragma unroll
    for (int off = 1; off < 16; off <<= 1) m = fmaxf(m, __shfl_xor(m, off, 32));
    float ssum = 0.f;
    #pragma unroll
    for (int nt = 0; nt < 5; ++nt) {
      bool ok = (nt * 16 + llo) < 77;
      float p = ok ? __expf(vals[nt] - m) : 0.f;
      vals[nt] = p;
      ssum += p;
    }
    #pragma unroll
    for (int off = 1; off < 16; off <<= 1) ssum += __shfl_xor(ssum, off, 32);
    float r = 1.0f / ssum;
    int row = j + (lhi << 3);
    #pragma unroll
    for (int nt = 0; nt < 5; ++nt)
      Ps[wave][row * LDP + nt * 16 + llo] = (bf16_t)(vals[nt] * r);
  }
  // per-wave private P region: LDS ops from one wave are in-order (DScnt)

  // ---- out = P V : 4 N-tiles (64 dims) x 3 K-steps (96 padded keys)
  v8f o[4] = {};
  #pragma unroll
  for (int ks = 0; ks < 3; ++ks) {
    const bf16_t* pp = &Ps[wave][llo * LDP + ks * 32 + (lhi << 3)];
    v16bf pf = cat8(*(const v8bf*)pp, *(const v8bf*)(pp + 16));
    #pragma unroll
    for (int nt = 0; nt < 4; ++nt) {
      const bf16_t* pv = &Vt[(nt * 16 + llo) * LDVT + ks * 32 + (lhi << 3)];
      v16bf vf = cat8(*(const v8bf*)pv, *(const v8bf*)(pv + 16));
      o[nt] = wmma_bf16(pf, vf, o[nt]);
    }
  }

  // ---- write attention output (bf16, head-interleaved like Q layout)
  bf16_t* obase = O + ((size_t)(b * Nq) + q0) * 1024 + h * 64;
  #pragma unroll
  for (int nt = 0; nt < 4; ++nt)
    #pragma unroll
    for (int j = 0; j < 8; ++j) {
      int row = j + (lhi << 3);
      obase[(size_t)row * 1024 + nt * 16 + llo] = (bf16_t)o[nt][j];
    }
}

// ---------------------------------------------------------------------------
extern "C" void kernel_launch(void* const* d_in, const int* in_sizes, int n_in,
                              void* d_out, int out_size, void* d_ws, size_t ws_size,
                              hipStream_t stream) {
  (void)in_sizes; (void)n_in; (void)out_size; (void)ws_size;
  const float* x       = (const float*)d_in[0];   // [8,4096,1024]
  const float* context = (const float*)d_in[1];   // [8,77,768]
  const float* Wq      = (const float*)d_in[2];   // [1024,1024]
  const float* Wk      = (const float*)d_in[3];   // [768,1024]
  const float* Wv      = (const float*)d_in[4];   // [768,1024]
  const float* Wo      = (const float*)d_in[5];   // [1024,1024]
  const float* bo      = (const float*)d_in[6];   // [1024]
  float* out = (float*)d_out;

  const int B = 8, Nq = 4096, Nk = 77, QD = 1024, CD = 768, ID = 1024;
  const int Mq = B * Nq;     // 32768
  const int Mk = B * Nk;     // 616

  // workspace: Q (bf16, reused in-place as attention output), K, V (bf16)
  bf16_t* qws = (bf16_t*)d_ws;                       // 32768*1024 bf16 = 64 MB
  bf16_t* kws = qws + (size_t)Mq * ID;               // 616*1024
  bf16_t* vws = kws + (size_t)Mk * ID;

  dim3 blk(256);
  // Q = x @ Wq  -> bf16
  gemm_bf16_wmma<float><<<dim3(ID / BN, Mq / BM), blk, 0, stream>>>(
      x, Wq, nullptr, nullptr, qws, Mq, ID, QD);
  // K = context @ Wk -> bf16
  gemm_bf16_wmma<float><<<dim3(ID / BN, (Mk + BM - 1) / BM), blk, 0, stream>>>(
      context, Wk, nullptr, nullptr, kws, Mk, ID, CD);
  // V = context @ Wv -> bf16
  gemm_bf16_wmma<float><<<dim3(ID / BN, (Mk + BM - 1) / BM), blk, 0, stream>>>(
      context, Wv, nullptr, nullptr, vws, Mk, ID, CD);
  // fused attention (writes in-place over Q buffer; per-block regions disjoint)
  attn_fused<<<dim3(Nq / 128, 16, B), blk, 0, stream>>>(qws, kws, vws, qws, Nq, Nk);
  // out = attn_out @ Wo + bo -> fp32
  gemm_bf16_wmma<bf16_t><<<dim3(QD / BN, Mq / BM), blk, 0, stream>>>(
      qws, Wo, bo, out, nullptr, Mq, QD, ID);
}
